// _NonLocalBlockND_54460185313414
// MI455X (gfx1250) — compile-verified, matrix operand order checked by reading, not run on hardware
//
#include <hip/hip_runtime.h>
#include <hip/hip_bf16.h>

typedef __attribute__((ext_vector_type(16))) __bf16 v16bf;
typedef __attribute__((ext_vector_type(8)))  __bf16 v8bf;
typedef __attribute__((ext_vector_type(8)))  float  v8f;
typedef __attribute__((ext_vector_type(4)))  unsigned u32x4;
typedef __attribute__((ext_vector_type(8)))  unsigned u32x8;

#define NB 16
#define NC 256
#define NCI 128
#define NN 4096   // 64*64
#define NM 1024   // 32*32

#define KROW 136  // padded K-chunk row stride in bf16 elems (128 + 8 pad)
#define VROW 40   // padded V-chunk row stride in bf16 elems (32 + 8 pad)

__device__ __forceinline__ v8f wmma_bf16(v16bf a, v16bf b, v8f c) {
    return __builtin_amdgcn_wmma_f32_16x16x32_bf16(false, a, false, b, (short)0, c, false, false);
}

__device__ __forceinline__ __bf16 f2bf(float f) {
    union { float f; unsigned u; } v; v.f = f;
    unsigned r = v.u + 0x7FFFu + ((v.u >> 16) & 1u);   // round to nearest even
    unsigned short h = (unsigned short)(r >> 16);
    union { unsigned short s; __bf16 b; } o; o.s = h; return o.b;
}
__device__ __forceinline__ float bf2f(__bf16 b) {
    union { unsigned short s; __bf16 b; } i; i.b = b;
    union { float f; unsigned u; } v; v.u = ((unsigned)i.s) << 16; return v.f;
}

// Load a 16x32 A (or 32x16 B) bf16 fragment: per-lane two contiguous 16B chunks
// at +0 (K 0..7 low half / 8..15 high half) and +16 elements (K 16..23 / 24..31).
__device__ __forceinline__ v16bf ldfrag(const __bf16* p) {
    v8bf lo = *(const v8bf*)p;
    v8bf hi = *(const v8bf*)(p + 16);
    v16bf r;
#pragma unroll
    for (int i = 0; i < 8; i++) { r[i] = lo[i]; r[8 + i] = hi[i]; }
    return r;
}

// ---------------------------------------------------------------------------
// TDM: async 2D bf16 tile load Global -> LDS with row padding (bank-conflict
// avoidance). D# per CDNA5 ISA ch.8: group0 = count/lds_addr/global_addr/type,
// group1 = data_size, dims, strides, pad config. Issued by one wave; tracked
// with TENSORcnt.
//   tile_d0: elements per row, tile_d1: rows, stride0: row stride (elements)
//   pad_int: code (pad every 2^(code+1) DWORDs), pad_amt: code (code+1 DWORDs)
// ---------------------------------------------------------------------------
__device__ __forceinline__ void tdm_load_2d(unsigned lds_off, const __bf16* gptr,
                                            unsigned tile_d0, unsigned tile_d1,
                                            unsigned stride0,
                                            unsigned pad_int, unsigned pad_amt) {
    unsigned long long ga = (unsigned long long)(size_t)gptr;
    u32x4 g0;
    g0[0] = 1u;                                        // count=1 (valid descriptor)
    g0[1] = lds_off;                                   // LDS byte address
    g0[2] = (unsigned)ga;                              // global_addr[31:0]
    g0[3] = (unsigned)(ga >> 32) | 0x80000000u;        // global_addr[56:32] | type=2
    u32x8 g1;
    g1[0] = 0x00010000u                                // data_size=1 (2 bytes)
          | (1u << 20)                                 // pad_enable
          | (pad_int << 22) | (pad_amt << 25);
    g1[1] = (stride0 & 0xFFFFu) << 16;                 // tensor_dim0[15:0] (=stride0)
    g1[2] = (stride0 >> 16) | ((tile_d1 & 0xFFFFu) << 16); // dim0 hi | tensor_dim1 lo
    g1[3] = (tile_d1 >> 16) | (tile_d0 << 16);         // tensor_dim1 hi | tile_dim0
    g1[4] = tile_d1;                                   // tile_dim1 (tile_dim2=0)
    g1[5] = stride0;                                   // tensor_dim0_stride[31:0]
    g1[6] = 0;
    g1[7] = 0;
    asm volatile("tensor_load_to_lds %0, %1" : : "s"(g0), "s"(g1) : "memory");
}

// ---------------------------------------------------------------------------
// Kernel 1: convert the four weight matrices to bf16 (concatenated in ws).
// ---------------------------------------------------------------------------
__global__ void wcvt_kernel(const float* __restrict__ wt, const float* __restrict__ wp,
                            const float* __restrict__ wg, const float* __restrict__ wo,
                            __bf16* __restrict__ out) {
    int i = blockIdx.x * 256 + threadIdx.x;          // 0..131071
    const float* src; int j = i;
    if (i < 32768)       { src = wt; }
    else if (i < 65536)  { src = wp; j = i - 32768; }
    else if (i < 98304)  { src = wg; j = i - 65536; }
    else                 { src = wo; j = i - 98304; }
    out[i] = f2bf(src[j]);
}

// ---------------------------------------------------------------------------
// Kernel 2: fused theta/phi/g 1x1-conv projections via WMMA.
// ---------------------------------------------------------------------------
__global__ __launch_bounds__(128) void proj_kernel(
    const float* __restrict__ x, const __bf16* __restrict__ wbf,
    const float* __restrict__ bth, const float* __restrict__ bph, const float* __restrict__ bg,
    __bf16* __restrict__ th, __bf16* __restrict__ ph, __bf16* __restrict__ gf) {
    __shared__ __bf16 xt[16 * 264];                  // [pixel][channel], padded rows
    int wgid = blockIdx.x;                            // B * (N/16)
    int b = wgid >> 8;
    int pbase = (wgid & 255) * 16;
    int t = threadIdx.x;
    const float* xb = x + (size_t)b * NC * NN;
#pragma unroll
    for (int i = 0; i < 32; i++) {
        int idx = i * 128 + t;                        // 4096 = 256ch * 16px
        int c = idx >> 4, p = idx & 15;
        xt[p * 264 + c] = f2bf(xb[(size_t)c * NN + pbase + p]);
    }
    __syncthreads();

    int lane = t & 31, wave = t >> 5;
    int hi = lane >> 4, lr = lane & 15;

    v16bf afr[8];                                     // A frags, shared by all tiles
#pragma unroll
    for (int kk = 0; kk < 8; kk++)
        afr[kk] = ldfrag(&xt[lr * 264 + kk * 32 + hi * 8]);

    for (int tt = 0; tt < 6; tt++) {                  // 24 output tiles / 4 waves
        int tile = wave * 6 + tt;
        int which = tile >> 3;                        // 0=theta 1=phi 2=g
        int cib = (tile & 7) * 16;
        const __bf16* w = wbf + which * 32768;
        const float* bias = which == 0 ? bth : (which == 1 ? bph : bg);
        __bf16* dst = which == 0 ? th : (which == 1 ? ph : gf);
        v16bf bfr[8];
#pragma unroll
        for (int kk = 0; kk < 8; kk++)                // cluster the 16 loads
            bfr[kk] = ldfrag(w + (size_t)(cib + lr) * NC + kk * 32 + hi * 8);
        v8f acc = {};
#pragma unroll
        for (int kk = 0; kk < 8; kk++)
            acc = wmma_bf16(afr[kk], bfr[kk], acc);
        float bv = bias[cib + lr];
        __bf16* drow = dst + ((size_t)b * NN + pbase) * NCI + cib;
#pragma unroll
        for (int v = 0; v < 8; v++) {                 // D: m = v + 8*hi, n = lr
            int m = v + 8 * hi;
            drow[(size_t)m * NCI + lr] = f2bf(acc[v] + bv);
        }
    }
}

// ---------------------------------------------------------------------------
// Kernel 3: 2x2 maxpool of phi -> [B, M, Ci] bf16 (key-major for B frags)
// ---------------------------------------------------------------------------
__global__ void pool_phi_kernel(const __bf16* __restrict__ ph, __bf16* __restrict__ php) {
    int i = blockIdx.x * 256 + threadIdx.x;           // B*M*Ci
    int ci = i & 127;
    int m = (i >> 7) & 1023;
    int b = i >> 17;
    int h2 = m >> 5, w2 = m & 31;
    int n0 = (h2 * 2) * 64 + w2 * 2;
    const __bf16* s = ph + ((size_t)b * NN + n0) * NCI + ci;
    float a = bf2f(s[0]);
    a = fmaxf(a, bf2f(s[NCI]));
    a = fmaxf(a, bf2f(s[64 * NCI]));
    a = fmaxf(a, bf2f(s[65 * NCI]));
    php[i] = f2bf(a);
}

// ---------------------------------------------------------------------------
// Kernel 4: 2x2 maxpool of g with transpose -> [B, Ci, M] bf16 (for V frags)
// ---------------------------------------------------------------------------
__global__ void pool_g_kernel(const __bf16* __restrict__ gf, __bf16* __restrict__ gt) {
    int i = blockIdx.x * 256 + threadIdx.x;           // b*Ci*M + ci*M + m
    int m = i & 1023;
    int ci = (i >> 10) & 127;
    int b = i >> 17;
    int h2 = m >> 5, w2 = m & 31;
    int n0 = (h2 * 2) * 64 + w2 * 2;
    const __bf16* s = gf + ((size_t)b * NN + n0) * NCI + ci;
    float a = bf2f(s[0]);
    a = fmaxf(a, bf2f(s[NCI]));
    a = fmaxf(a, bf2f(s[64 * NCI]));
    a = fmaxf(a, bf2f(s[65 * NCI]));
    gt[i] = f2bf(a);
}

// ---------------------------------------------------------------------------
// Kernel 5: attention with TDM-streamed K/V. Per workgroup: 4 waves x 16
// queries share double-buffered K [32x128] / V [128x32] bf16 chunks in LDS
// (loaded by tensor_load_to_lds with row padding, synced via TENSORcnt +
// workgroup barrier). Two-pass softmax; all GEMMs via bf16 WMMA.
// ---------------------------------------------------------------------------
__global__ __launch_bounds__(128) void attn_kernel(
    const __bf16* __restrict__ th, const __bf16* __restrict__ php,
    const __bf16* __restrict__ gt, __bf16* __restrict__ yb) {
    __shared__ __bf16 kbuf[2][32 * KROW];             // 2 x 8704 B
    __shared__ __bf16 vbuf[2][128 * VROW];            // 2 x 10240 B
    __shared__ __bf16 pb[4][16 * 32];                 // per-wave P tile

    int wgid = blockIdx.x;                            // B * (N/64)
    int b = wgid >> 6;
    int wave = threadIdx.x >> 5;
    int qb = (wgid & 63) * 64 + wave * 16;
    int lane = threadIdx.x & 31, hi = lane >> 4, lr = lane & 15;
    const __bf16* Q = th + ((size_t)b * NN + qb) * NCI;
    const __bf16* K = php + (size_t)b * NM * NCI;
    const __bf16* V = gt + (size_t)b * NCI * NM;
    __bf16* pbw = pb[wave];
    bool issuer = (wave == 0);
    unsigned koff[2] = { (unsigned)(size_t)&kbuf[0][0], (unsigned)(size_t)&kbuf[1][0] };
    unsigned voff[2] = { (unsigned)(size_t)&vbuf[0][0], (unsigned)(size_t)&vbuf[1][0] };

    v16bf qf[4];
#pragma unroll
    for (int kk = 0; kk < 4; kk++)
        qf[kk] = ldfrag(Q + (size_t)lr * NCI + kk * 32 + hi * 8);

    // ---- pass 1: row maxima (stream K chunks of 32 keys) ----
    float rmax[8];
#pragma unroll
    for (int v = 0; v < 8; v++) rmax[v] = -1e30f;

    if (issuer) tdm_load_2d(koff[0], K, 128, 32, 128, 5, 3);
    for (int c = 0; c < 32; c++) {
        if (issuer) __builtin_amdgcn_s_wait_tensorcnt(0);
        __syncthreads();                              // publish chunk c
        if (issuer && c < 31)
            tdm_load_2d(koff[(c + 1) & 1], K + (size_t)(c + 1) * 32 * NCI, 128, 32, 128, 5, 3);
        const __bf16* kb = kbuf[c & 1];
#pragma unroll
        for (int st = 0; st < 2; st++) {
            v8f s = {};
#pragma unroll
            for (int kk = 0; kk < 4; kk++) {
                v16bf kf = ldfrag(kb + (size_t)(st * 16 + lr) * KROW + kk * 32 + hi * 8);
                s = wmma_bf16(qf[kk], kf, s);
            }
#pragma unroll
            for (int v = 0; v < 8; v++) {             // reduce across 16 cols (half-wave)
                float xv = s[v];
                xv = fmaxf(xv, __shfl_xor(xv, 1, 32));
                xv = fmaxf(xv, __shfl_xor(xv, 2, 32));
                xv = fmaxf(xv, __shfl_xor(xv, 4, 32));
                xv = fmaxf(xv, __shfl_xor(xv, 8, 32));
                rmax[v] = fmaxf(rmax[v], xv);
            }
        }
        __syncthreads();                              // all reads of chunk c done
    }

    // ---- pass 2: exp, row sums, P@V (stream K and V chunks) ----
    float rsum[8];
    v8f acc[8];
    v8f zero = {};
#pragma unroll
    for (int v = 0; v < 8; v++) { rsum[v] = 0.f; acc[v] = zero; }

    if (issuer) {
        tdm_load_2d(koff[0], K, 128, 32, 128, 5, 3);
        tdm_load_2d(voff[0], V, 32, 128, 1024, 3, 3);
    }
    for (int c = 0; c < 32; c++) {
        if (issuer) __builtin_amdgcn_s_wait_tensorcnt(0);
        __syncthreads();                              // publish chunk c
        if (issuer && c < 31) {
            tdm_load_2d(koff[(c + 1) & 1], K + (size_t)(c + 1) * 32 * NCI, 128, 32, 128, 5, 3);
            tdm_load_2d(voff[(c + 1) & 1], V + (c + 1) * 32, 32, 128, 1024, 3, 3);
        }
        const __bf16* kb = kbuf[c & 1];
        const __bf16* vb = vbuf[c & 1];
#pragma unroll
        for (int st = 0; st < 2; st++) {
            v8f s = {};
#pragma unroll
            for (int kk = 0; kk < 4; kk++) {
                v16bf kf = ldfrag(kb + (size_t)(st * 16 + lr) * KROW + kk * 32 + hi * 8);
                s = wmma_bf16(qf[kk], kf, s);
            }
#pragma unroll
            for (int v = 0; v < 8; v++) {
                float p = __expf(s[v] - rmax[v]);
                float ss = p;
                ss += __shfl_xor(ss, 1, 32);
                ss += __shfl_xor(ss, 2, 32);
                ss += __shfl_xor(ss, 4, 32);
                ss += __shfl_xor(ss, 8, 32);
                rsum[v] += ss;
                pbw[(v + 8 * hi) * 32 + st * 16 + lr] = f2bf(p);
            }
        }
        __syncthreads();                              // order P stores vs vector reads
        v16bf pf = ldfrag(pbw + lr * 32 + hi * 8);    // P as A fragment [16q x 32k]
#pragma unroll
        for (int ct = 0; ct < 8; ct++) {              // Ci = 8 tiles of 16
            v16bf vf = ldfrag(vb + (size_t)(ct * 16 + lr) * VROW + hi * 8);
            acc[ct] = wmma_bf16(pf, vf, acc[ct]);
        }
        __syncthreads();                              // chunk c fully consumed
    }

    __bf16* Y = yb + ((size_t)b * NN + qb) * NCI;
#pragma unroll
    for (int ct = 0; ct < 8; ct++) {
#pragma unroll
        for (int v = 0; v < 8; v++) {
            int m = v + 8 * hi;
            Y[(size_t)m * NCI + ct * 16 + lr] = f2bf(acc[ct][v] / rsum[v]);
        }
    }
}

// ---------------------------------------------------------------------------
// Kernel 6: output 1x1 conv + bias + residual (lane-contiguous f32 stores).
// ---------------------------------------------------------------------------
__global__ __launch_bounds__(128) void out_kernel(
    const __bf16* __restrict__ yb, const __bf16* __restrict__ wob,
    const float* __restrict__ bo, const float* __restrict__ x,
    float* __restrict__ out) {
    int wgid = blockIdx.x;                            // B * (N/64)
    int b = wgid >> 6;
    int pbase = (wgid & 63) * 64 + (threadIdx.x >> 5) * 16;
    int lane = threadIdx.x & 31, hi = lane >> 4, lr = lane & 15;
    const __bf16* Y = yb + (size_t)b * NN * NCI;

    v16bf yf[4];                                      // B frags hoisted across c-tiles
#pragma unroll
    for (int kk = 0; kk < 4; kk++)
        yf[kk] = ldfrag(Y + (size_t)(pbase + lr) * NCI + kk * 32 + hi * 8);

    for (int ctile = 0; ctile < 16; ctile++) {
        int cb = ctile * 16;
        v16bf af[4];
#pragma unroll
        for (int kk = 0; kk < 4; kk++)
            af[kk] = ldfrag(wob + (size_t)(cb + lr) * NCI + kk * 32 + hi * 8);
        v8f acc = {};
#pragma unroll
        for (int kk = 0; kk < 4; kk++)
            acc = wmma_bf16(af[kk], yf[kk], acc);
#pragma unroll
        for (int v = 0; v < 8; v++) {
            int c = cb + v + 8 * hi;
            size_t idx = ((size_t)b * NC + c) * NN + pbase + lr;
            out[idx] = acc[v] + bo[c] + x[idx];
        }
    }
}

// ---------------------------------------------------------------------------
extern "C" void kernel_launch(void* const* d_in, const int* in_sizes, int n_in,
                              void* d_out, int out_size, void* d_ws, size_t ws_size,
                              hipStream_t stream) {
    (void)in_sizes; (void)n_in; (void)out_size; (void)ws_size;
    const float* x   = (const float*)d_in[0];
    const float* wt  = (const float*)d_in[1];
    const float* bt  = (const float*)d_in[2];
    const float* wp  = (const float*)d_in[3];
    const float* bp  = (const float*)d_in[4];
    const float* wg  = (const float*)d_in[5];
    const float* bg  = (const float*)d_in[6];
    const float* wo  = (const float*)d_in[7];
    const float* bo  = (const float*)d_in[8];
    float* out = (float*)d_out;

    char* ws = (char*)d_ws;
    __bf16* wcat = (__bf16*)ws;                               // 4*32768 bf16
    __bf16* wob  = wcat + 98304;                              // w_out bf16
    __bf16* thb  = (__bf16*)(ws + 262144);                    // [B,N,Ci]
    __bf16* phb  = (__bf16*)(ws + 17039360);                  // [B,N,Ci]
    __bf16* gfb  = (__bf16*)(ws + 33816576);                  // [B,N,Ci]
    __bf16* phpb = (__bf16*)(ws + 50593792);                  // [B,M,Ci]
    __bf16* gtb  = (__bf16*)(ws + 54788096);                  // [B,Ci,M]
    __bf16* ybb  = (__bf16*)(ws + 58982400);                  // [B,N,Ci]

    wcvt_kernel<<<512, 256, 0, stream>>>(wt, wp, wg, wo, wcat);
    proj_kernel<<<NB * (NN / 16), 128, 0, stream>>>(x, wcat, bt, bp, bg, thb, phb, gfb);
    pool_phi_kernel<<<(NB * NM * NCI) / 256, 256, 0, stream>>>(phb, phpb);
    pool_g_kernel<<<(NB * NCI * NM) / 256, 256, 0, stream>>>(gfb, gtb);
    attn_kernel<<<NB * (NN / 64), 128, 0, stream>>>(thb, phpb, gtb, ybb);
    out_kernel<<<NB * (NN / 64), 128, 0, stream>>>(ybb, wob, bo, x, out);
}
